// GCN_56882546868697
// MI455X (gfx1250) — matrix operand order repaired and checked by reading, not compiled, not run on hardware
//
#include <hip/hip_runtime.h>

typedef __attribute__((ext_vector_type(16))) _Float16 v16h;
typedef __attribute__((ext_vector_type(8)))  float    v8f;

#define D_FEAT 64
#define LATENT 64
#define NCLS   16
#define NEG_SLOPE 0.01f

#define WMMA_F16(A, B, C) \
  __builtin_amdgcn_wmma_f32_16x16x32_f16(false, (A), false, (B), (short)0, (C), false, false)

// ---------------------------------------------------------------- utilities
__global__ void GCN_zero_kernel(float* __restrict__ p, int n) {
  int i = blockIdx.x * blockDim.x + threadIdx.x;
  if (i < n) p[i] = 0.0f;
}

__global__ void GCN_deg_kernel(const int* __restrict__ snd, const int* __restrict__ rcv,
                               float* __restrict__ degs, float* __restrict__ degr, int nE) {
  int e = blockIdx.x * blockDim.x + threadIdx.x;
  if (e < nE) {
    atomicAdd(&degs[snd[e]], 1.0f);
    atomicAdd(&degr[rcv[e]], 1.0f);
  }
}

__global__ void GCN_rsqrt_kernel(float* __restrict__ d, int n) {
  int i = blockIdx.x * blockDim.x + threadIdx.x;
  if (i < n) d[i] = rsqrtf(fmaxf(d[i], 1.0f));
}

// ------------------------------------------------------------- weight pack
// Pre-swizzle W (f32, row-major [K x N], ldw floats/row) into f16 in the exact
// per-lane WMMA B-operand register layout, so each GEMM wave fetches a whole
// 32x16 B operand as one contiguous 32-byte vector load per lane (2x b128).
__global__ void GCN_packB_kernel(const float* __restrict__ W, _Float16* __restrict__ Wp,
                                 int ldw, int nTilesN) {
  int i = blockIdx.x * blockDim.x + threadIdx.x;
  int total = nTilesN * 2 * 32 * 16;
  if (i >= total) return;
  int elem = i & 15;
  int lane = (i >> 4) & 31;
  int kk   = (i >> 9) & 1;
  int tN   = i >> 10;
  int k = kk * 32 + (elem >> 3) * 16 + (lane >> 4) * 8 + (elem & 7);
  int n = tN * 16 + (lane & 15);
  Wp[i] = (_Float16)W[k * ldw + n];
}

// --------------------------------------------- GEMM1: hs = (X@W0 + b0) * invs[row]
// One wave per PAIR of 16x16 output tiles (same N-tile, adjacent M-tiles):
// 4 v_wmma in two independent accumulator chains, interleaved to hide the
// WMMA->WMMA RAW hazard and reuse the B registers across both tiles.
__global__ void GCN_gemm1_wmma(const float* __restrict__ X, const v16h* __restrict__ Wp,
                               const float* __restrict__ bias, const float* __restrict__ invs,
                               float* __restrict__ hs, int nTilesM) {
  const int wave = (blockIdx.x * blockDim.x + threadIdx.x) >> 5;
  const int lane = threadIdx.x & 31;
  const int pair = wave >> 2;          // LATENT/16 == 4 N-tiles
  const int tN   = wave & 3;
  const int tM0  = pair * 2;
  if (tM0 >= nTilesM) return;          // wave-uniform branch (EXEC all-ones for WMMA)
  const int tM1  = tM0 + 1;
  const bool has1 = (tM1 < nTilesM);   // nTilesM is even for this problem

  const int nloc = lane & 15;
  const int hi   = lane >> 4;          // 0 or 1
  const int kb   = hi * 8;             // A-matrix K sub-block per lane half
  const int n    = tN * 16 + nloc;

  // B operands: one 32B vector load each (pre-swizzled f16), shared by both tiles
  const v16h b0v = Wp[(tN * 2 + 0) * 32 + lane];
  const v16h b1v = Wp[(tN * 2 + 1) * 32 + lane];

  // A operands for both M-tiles (clamp second tile to keep straight-line code)
  const float* ar0 = X + (size_t)(tM0 * 16 + nloc) * D_FEAT;
  const float* ar1 = has1 ? (ar0 + 16 * D_FEAT) : ar0;
  v16h a00, a01, a10, a11;
#pragma unroll
  for (int i = 0; i < 8; ++i) {        // A layout: [0..7]=K kb.., [8..15]=K 16+kb..
    a00[i]     = (_Float16)ar0[kb + i];
    a00[8 + i] = (_Float16)ar0[16 + kb + i];
    a01[i]     = (_Float16)ar0[32 + kb + i];
    a01[8 + i] = (_Float16)ar0[48 + kb + i];
    a10[i]     = (_Float16)ar1[kb + i];
    a10[8 + i] = (_Float16)ar1[16 + kb + i];
    a11[i]     = (_Float16)ar1[32 + kb + i];
    a11[8 + i] = (_Float16)ar1[48 + kb + i];
  }

  const float bn = bias[n];
  v8f c0, c1;
#pragma unroll
  for (int v = 0; v < 8; ++v) { c0[v] = bn; c1[v] = bn; }  // fused bias

  // interleaved independent chains: adjacent WMMAs have no register overlap
  c0 = WMMA_F16(a00, b0v, c0);
  c1 = WMMA_F16(a10, b0v, c1);
  c0 = WMMA_F16(a01, b1v, c0);
  c1 = WMMA_F16(a11, b1v, c1);

#pragma unroll
  for (int v = 0; v < 8; ++v) {        // C/D layout: VGPR v -> M = v + 8*hi
    int m = tM0 * 16 + v + hi * 8;
    hs[(size_t)m * LATENT + n] = c0[v] * invs[m];
  }
  if (has1) {
#pragma unroll
    for (int v = 0; v < 8; ++v) {
      int m = tM1 * 16 + v + hi * 8;
      hs[(size_t)m * LATENT + n] = c1[v] * invs[m];
    }
  }
}

// ------------------------------- scatter layer 1: h[rcv] += hs[snd], 64 floats/edge
// 16 lanes per edge; each lane moves a float4 (one coalesced 256B b128 gather/edge).
__global__ void GCN_scatter64(const float* __restrict__ hs, const int* __restrict__ snd,
                              const int* __restrict__ rcv, float* __restrict__ h, int nE) {
  const int t = blockIdx.x * blockDim.x + threadIdx.x;
  const int e = t >> 4;
  const int q = t & 15;
  if (e >= nE) return;
  const int ep = e + 128;                  // lookahead prefetch -> global_prefetch_b8
  if (ep < nE) __builtin_prefetch(hs + (size_t)snd[ep] * LATENT + q * 4, 0, 1);
  const int s = snd[e];
  const int r = rcv[e];
  const float4 v = *(const float4*)(hs + (size_t)s * LATENT + q * 4);
  float* dst = h + (size_t)r * LATENT + q * 4;
  atomicAdd(dst + 0, v.x);
  atomicAdd(dst + 1, v.y);
  atomicAdd(dst + 2, v.z);
  atomicAdd(dst + 3, v.w);
}

// -------- GEMM2: z = sigmoid(leakyrelu(h * invr[row]) @ W1 + b1); fused epilogues
// Same two-M-tile / dual-chain structure as GEMM1 (single N-tile: 16 classes).
__global__ void GCN_gemm2_wmma(const float* __restrict__ H, const v16h* __restrict__ Wp,
                               const float* __restrict__ bias, const float* __restrict__ invr,
                               float* __restrict__ z, int nTilesM) {
  const int wave = (blockIdx.x * blockDim.x + threadIdx.x) >> 5;
  const int lane = threadIdx.x & 31;
  const int tM0 = wave * 2;
  if (tM0 >= nTilesM) return;
  const int tM1 = tM0 + 1;
  const bool has1 = (tM1 < nTilesM);

  const int nloc = lane & 15;
  const int hi   = lane >> 4;
  const int kb   = hi * 8;

  const v16h b0v = Wp[lane];
  const v16h b1v = Wp[32 + lane];

  const int row0 = tM0 * 16 + nloc;
  const int row1 = has1 ? (row0 + 16) : row0;
  const float sc0 = invr[row0];
  const float sc1 = invr[row1];
  const float* hr0 = H + (size_t)row0 * LATENT;
  const float* hr1 = H + (size_t)row1 * LATENT;

  v16h a00, a01, a10, a11;
#pragma unroll
  for (int i = 0; i < 8; ++i) {            // fused recv-degree scale + leaky-relu
    float x;
    x = hr0[kb + i] * sc0;      a00[i]     = (_Float16)(x >= 0.0f ? x : NEG_SLOPE * x);
    x = hr0[16 + kb + i] * sc0; a00[8 + i] = (_Float16)(x >= 0.0f ? x : NEG_SLOPE * x);
    x = hr0[32 + kb + i] * sc0; a01[i]     = (_Float16)(x >= 0.0f ? x : NEG_SLOPE * x);
    x = hr0[48 + kb + i] * sc0; a01[8 + i] = (_Float16)(x >= 0.0f ? x : NEG_SLOPE * x);
    x = hr1[kb + i] * sc1;      a10[i]     = (_Float16)(x >= 0.0f ? x : NEG_SLOPE * x);
    x = hr1[16 + kb + i] * sc1; a10[8 + i] = (_Float16)(x >= 0.0f ? x : NEG_SLOPE * x);
    x = hr1[32 + kb + i] * sc1; a11[i]     = (_Float16)(x >= 0.0f ? x : NEG_SLOPE * x);
    x = hr1[48 + kb + i] * sc1; a11[8 + i] = (_Float16)(x >= 0.0f ? x : NEG_SLOPE * x);
  }

  const float bn = bias[nloc];
  v8f c0, c1;
#pragma unroll
  for (int v = 0; v < 8; ++v) { c0[v] = bn; c1[v] = bn; }

  c0 = WMMA_F16(a00, b0v, c0);
  c1 = WMMA_F16(a10, b0v, c1);
  c0 = WMMA_F16(a01, b1v, c0);
  c1 = WMMA_F16(a11, b1v, c1);

#pragma unroll
  for (int v = 0; v < 8; ++v) {
    int m = tM0 * 16 + v + hi * 8;
    z[(size_t)m * NCLS + nloc] = 1.0f / (1.0f + __expf(-c0[v]));  // fused sigmoid
  }
  if (has1) {
#pragma unroll
    for (int v = 0; v < 8; ++v) {
      int m = tM1 * 16 + v + hi * 8;
      z[(size_t)m * NCLS + nloc] = 1.0f / (1.0f + __expf(-c1[v]));
    }
  }
}

// -------------------------- scatter layer 2: out[rcv] += z[snd], 16 floats/edge
__global__ void GCN_scatter16(const float* __restrict__ z, const int* __restrict__ snd,
                              const int* __restrict__ rcv, float* __restrict__ out, int nE) {
  const int t = blockIdx.x * blockDim.x + threadIdx.x;
  const int e = t >> 4;
  const int f = t & 15;
  if (e >= nE) return;
  atomicAdd(&out[(size_t)rcv[e] * NCLS + f], z[(size_t)snd[e] * NCLS + f]);
}

// ---------------------------------------------------------------------------
extern "C" void kernel_launch(void* const* d_in, const int* in_sizes, int n_in,
                              void* d_out, int out_size, void* d_ws, size_t ws_size,
                              hipStream_t stream) {
  const float* nodes = (const float*)d_in[0];
  const int*   snd   = (const int*)d_in[1];
  const int*   rcv   = (const int*)d_in[2];
  const float* W0    = (const float*)d_in[3];
  const float* b0    = (const float*)d_in[4];
  const float* W1    = (const float*)d_in[5];
  const float* b1    = (const float*)d_in[6];
  float*       out   = (float*)d_out;

  const int nNodes  = in_sizes[0] / D_FEAT;   // 100000
  const int nE      = in_sizes[1];            // 1600000
  const int nTilesM = (nNodes + 15) / 16;     // 6250 (100000 % 16 == 0)
  const int nPairs  = (nTilesM + 1) / 2;      // 3125

  // workspace carve-out (256B aligned)
  char* ws = (char*)d_ws;
  size_t off = 0;
  auto carve = [&](size_t bytes) -> void* {
    void* p = (void*)(ws + off);
    off = (off + bytes + 255) & ~(size_t)255;
    return p;
  };
  float*     deg  = (float*)carve((size_t)2 * nNodes * sizeof(float)); // [invs|invr]
  float*     invs = deg;
  float*     invr = deg + nNodes;
  float*     hs   = (float*)carve((size_t)nNodes * LATENT * sizeof(float));
  float*     h    = (float*)carve((size_t)nNodes * LATENT * sizeof(float));
  float*     z    = (float*)carve((size_t)nNodes * NCLS * sizeof(float));
  _Float16*  Wp0  = (_Float16*)carve((size_t)4 * 2 * 32 * 16 * sizeof(_Float16)); // 8KB
  _Float16*  Wp1  = (_Float16*)carve((size_t)1 * 2 * 32 * 16 * sizeof(_Float16)); // 2KB
  (void)ws_size; (void)n_in;

  const int B = 256;

  // zero accumulators + output (ws/d_out are poisoned by the harness)
  GCN_zero_kernel<<<(2 * nNodes + B - 1) / B, B, 0, stream>>>(deg, 2 * nNodes);
  GCN_zero_kernel<<<(nNodes * LATENT + B - 1) / B, B, 0, stream>>>(h, nNodes * LATENT);
  GCN_zero_kernel<<<(out_size + B - 1) / B, B, 0, stream>>>(out, out_size);

  // pre-swizzle weights into f16 WMMA B-register layout (one-off, tiny)
  GCN_packB_kernel<<<(4 * 2 * 32 * 16 + B - 1) / B, B, 0, stream>>>(W0, Wp0, LATENT, 4);
  GCN_packB_kernel<<<(1 * 2 * 32 * 16 + B - 1) / B, B, 0, stream>>>(W1, Wp1, NCLS, 1);

  // degrees -> rsqrt(max(deg,1))
  GCN_deg_kernel<<<(nE + B - 1) / B, B, 0, stream>>>(snd, rcv, invs, invr, nE);
  GCN_rsqrt_kernel<<<(2 * nNodes + B - 1) / B, B, 0, stream>>>(deg, 2 * nNodes);

  // layer 1: WMMA GEMM (bias + sender-degree scale fused), then edge scatter
  GCN_gemm1_wmma<<<(nPairs * 4 + 7) / 8, B, 0, stream>>>(nodes, (const v16h*)Wp0, b0, invs, hs, nTilesM);
  GCN_scatter64<<<(nE * 16 + B - 1) / B, B, 0, stream>>>(hs, snd, rcv, h, nE);

  // layer 2: WMMA GEMM (recv scale + leaky-relu + sigmoid fused), then scatter
  GCN_gemm2_wmma<<<(nPairs + 7) / 8, B, 0, stream>>>(h, (const v16h*)Wp1, b1, invr, z, nTilesM);
  GCN_scatter16<<<(nE * 16 + B - 1) / B, B, 0, stream>>>(z, snd, rcv, out, nE);
}
